// CNF_20985210208709
// MI455X (gfx1250) — matrix operand order repaired and checked by reading, not compiled, not run on hardware
//
#include <hip/hip_runtime.h>
#include <hip/hip_bf16.h>

typedef __attribute__((ext_vector_type(16))) __bf16 v16bf;
typedef __attribute__((ext_vector_type(8)))  __bf16 v8bf;
typedef __attribute__((ext_vector_type(8)))  float  v8f;

#define D_DIM  64
#define H_DIM  512
#define MBLK   16
#define NSTEPS 8

__device__ __forceinline__ v16bf cat8(v8bf lo, v8bf hi) {
  return __builtin_shufflevector(lo, hi, 0,1,2,3,4,5,6,7,8,9,10,11,12,13,14,15);
}
__device__ __forceinline__ v8f wmma_bf16(v16bf a, v16bf b, v8f c) {
  // D = A(16x32 bf16) * B(32x16 bf16) + C(16x16 f32)
  return __builtin_amdgcn_wmma_f32_16x16x32_bf16(false, a, false, b, (short)0, c,
                                                 false, false);
}

// Branch-free fast tanh: 1 - 2/(exp(2x)+1).
// exp overflow -> +inf -> result 1; underflow -> 0 -> result -1. No clamping needed.
__device__ __forceinline__ float tanh_fast(float x) {
  float e = __expf(2.0f * x);                    // v_exp_f32 (+ scale)
  return 1.0f - 2.0f * __builtin_amdgcn_rcpf(e + 1.0f);
}

// ---------------------------------------------------------------------------
// Prep 1: bf16 transposed copies of W1x (64x512 -> [n][k]=W1[k,n], 512x64)
//         and W3 (512x64 -> [n][k]=W3[k,n], 64x512)
// ---------------------------------------------------------------------------
__global__ void cnf_prep_small(const float* __restrict__ W1,
                               const float* __restrict__ W3,
                               __bf16* __restrict__ W1xT,
                               __bf16* __restrict__ W3T) {
  int i = blockIdx.x * 256 + threadIdx.x;
  if (i < H_DIM * D_DIM) {
    int n = i >> 6, k = i & 63;              // W1xT[n][k] = W1[k*512+n]
    W1xT[i] = (__bf16)W1[k * H_DIM + n];
    int n2 = i >> 9, k2 = i & 511;           // W3T[n2][k2] = W3[k2*64+n2]
    W3T[i] = (__bf16)W3[k2 * D_DIM + n2];
  }
}

// ---------------------------------------------------------------------------
// Prep 2: W2T[n][j] = bf16(W2[j,n]);  ET[n][j] = bf16(W2[j,n] * C[j,n])
//         C[j,n] = sum_o W1[o,j] * W3[n,o]   (trace-collapse matrix)
// ---------------------------------------------------------------------------
__global__ __launch_bounds__(512)
void cnf_prep_e(const float* __restrict__ W1, const float* __restrict__ W2,
                const float* __restrict__ W3,
                __bf16* __restrict__ W2T, __bf16* __restrict__ ET) {
  __shared__ float w3row[D_DIM];
  const int n = blockIdx.x;       // hidden-2 index (column of W2)
  const int j = threadIdx.x;      // hidden-1 index (row of W2)
  if (j < D_DIM) w3row[j] = W3[n * D_DIM + j];
  __syncthreads();
  float c = 0.f;
#pragma unroll 8
  for (int o = 0; o < D_DIM; ++o) c += W1[o * H_DIM + j] * w3row[o];
  float w2 = W2[j * H_DIM + n];
  W2T[n * H_DIM + j] = (__bf16)w2;
  ET[n * H_DIM + j]  = (__bf16)(w2 * c);
}

// ---------------------------------------------------------------------------
// Main kernel: one workgroup integrates 16 samples through all 8 RK4 steps.
// 8 wave32 per block; each wave owns 4 column tiles (64 of 512 N-columns).
// ---------------------------------------------------------------------------
__global__ __launch_bounds__(256)
void cnf_main(const float* __restrict__ x_in,
              const float* __restrict__ W1,  // fp32 (only the t-row is used)
              const float* __restrict__ b1,
              const float* __restrict__ b2,
              const float* __restrict__ b3,
              const __bf16* __restrict__ W1xT,  // [512][64]
              const __bf16* __restrict__ W2T,   // [512][512]
              const __bf16* __restrict__ ET,    // [512][512]
              const __bf16* __restrict__ W3T,   // [64][512]
              float* __restrict__ z_out, float* __restrict__ ld_out) {
  __shared__ __bf16 xsb[MBLK][D_DIM];   // stage input, bf16 (A for layer 1)
  __shared__ __bf16 h1b[MBLK][H_DIM];   // tanh(a1)            (A for layer 2)
  __shared__ __bf16 d1b[MBLK][H_DIM];   // 1 - h1^2            (A for G)
  __shared__ __bf16 h2b[MBLK][H_DIM];   // tanh(a2)            (A for layer 3)
  __shared__ float  xcur[MBLK][D_DIM];
  __shared__ float  xacc[MBLK][D_DIM];
  __shared__ float  dxo [MBLK][D_DIM];
  __shared__ float  trred[MBLK];
  __shared__ float  ldcur[MBLK];
  __shared__ float  ldacc[MBLK];

  const int tid  = threadIdx.x;
  const int lane = tid & 31;
  const int wave = tid >> 5;
  const int rowbase = blockIdx.x * MBLK;

  const int arow  = lane & 15;                 // A-fragment row
  const int koff  = (lane < 16) ? 0 : 8;       // A-fragment K sub-offset
  const int kboff = (lane < 16) ? 0 : 8;       // B-fragment K sub-offset
  const int mbase = (lane < 16) ? 0 : 8;       // C-fragment row base
  const int ncl   = lane & 15;                 // C/B column within tile

  // Per-thread loop-invariant column constants (kept in VGPRs).
  float w1t_r[4], b1_r[4], b2_r[4];
#pragma unroll
  for (int j = 0; j < 4; ++j) {
    const int ncol = (wave * 4 + j) * 16 + ncl;
    w1t_r[j] = W1[D_DIM * H_DIM + ncol];
    b1_r[j]  = b1[ncol];
    b2_r[j]  = b2[ncol];
  }
  const float b3_r = (wave < 4) ? b3[wave * 16 + ncl] : 0.f;

  for (int e = tid; e < MBLK * D_DIM; e += 256) {
    int m = e >> 6, k = e & 63;
    float v = x_in[(rowbase + m) * D_DIM + k];
    xcur[m][k] = v;
    xsb[m][k]  = (__bf16)v;
  }
  if (tid < MBLK) { ldcur[tid] = 0.f; ldacc[tid] = 0.f; trred[tid] = 0.f; }
  __syncthreads();

  const float dt = 1.0f / NSTEPS;

  for (int step = 0; step < NSTEPS; ++step) {
    const float t0 = (float)step * dt;
    for (int stage = 0; stage < 4; ++stage) {
      const float ts  = t0 + ((stage == 0) ? 0.f : (stage == 3) ? dt : 0.5f * dt);
      const float wgt = (stage == 1 || stage == 2) ? 2.f : 1.f;

      float d2reg[4][8];   // lives layer2-post -> G-post (WMMA C layouts match)

      // ---------------- layer 1: A1 = xs @ W1x  (M=16,N=512,K=64) ----------
      {
        v8bf l0 = *(const v8bf*)&xsb[arow][0 + koff];
        v8bf h0 = *(const v8bf*)&xsb[arow][16 + koff];
        v8bf l1 = *(const v8bf*)&xsb[arow][32 + koff];
        v8bf h1 = *(const v8bf*)&xsb[arow][48 + koff];
        v16bf a0 = cat8(l0, h0), a1 = cat8(l1, h1);
#pragma unroll
        for (int j = 0; j < 4; ++j) {
          const int ncol = (wave * 4 + j) * 16 + ncl;
          v8f acc = {};
          {
            const v8bf* p = (const v8bf*)&W1xT[ncol * D_DIM + kboff];
            acc = wmma_bf16(a0, cat8(p[0], p[2]), acc);
          }
          {
            const v8bf* p = (const v8bf*)&W1xT[ncol * D_DIM + 32 + kboff];
            acc = wmma_bf16(a1, cat8(p[0], p[2]), acc);
          }
          const float add = ts * w1t_r[j] + b1_r[j];
#pragma unroll
          for (int r = 0; r < 8; ++r) {
            float h = tanh_fast(acc[r] + add);
            h1b[mbase + r][ncol] = (__bf16)h;
            d1b[mbase + r][ncol] = (__bf16)(1.f - h * h);
          }
        }
      }
      __syncthreads();

      // ---------------- layer 2: A2 = H1 @ W2  (M=16,N=512,K=512) ----------
      {
        v8f acc[4] = {v8f{}, v8f{}, v8f{}, v8f{}};
        for (int kb = 0; kb < H_DIM; kb += 32) {
          v8bf lo = *(const v8bf*)&h1b[arow][kb + koff];
          v8bf hi = *(const v8bf*)&h1b[arow][kb + 16 + koff];
          v16bf a = cat8(lo, hi);
#pragma unroll
          for (int j = 0; j < 4; ++j) {
            const int ncol = (wave * 4 + j) * 16 + ncl;
            const v8bf* p = (const v8bf*)&W2T[ncol * H_DIM + kb + kboff];
            acc[j] = wmma_bf16(a, cat8(p[0], p[2]), acc[j]);
          }
        }
#pragma unroll
        for (int j = 0; j < 4; ++j) {
          const int ncol = (wave * 4 + j) * 16 + ncl;
#pragma unroll
          for (int r = 0; r < 8; ++r) {
            float h = tanh_fast(acc[j][r] + b2_r[j]);
            h2b[mbase + r][ncol] = (__bf16)h;
            d2reg[j][r] = 1.f - h * h;
          }
        }
      }
      __syncthreads();

      // ---------------- G = D1 @ E, fused trace:  tr_m = sum_n G*d2 --------
      {
        v8f acc[4] = {v8f{}, v8f{}, v8f{}, v8f{}};
        for (int kb = 0; kb < H_DIM; kb += 32) {
          v8bf lo = *(const v8bf*)&d1b[arow][kb + koff];
          v8bf hi = *(const v8bf*)&d1b[arow][kb + 16 + koff];
          v16bf a = cat8(lo, hi);
#pragma unroll
          for (int j = 0; j < 4; ++j) {
            const int ncol = (wave * 4 + j) * 16 + ncl;
            const v8bf* p = (const v8bf*)&ET[ncol * H_DIM + kb + kboff];
            acc[j] = wmma_bf16(a, cat8(p[0], p[2]), acc[j]);
          }
        }
#pragma unroll
        for (int r = 0; r < 8; ++r) {
          float part = 0.f;
#pragma unroll
          for (int j = 0; j < 4; ++j) part += acc[j][r] * d2reg[j][r];
          part += __shfl_xor(part, 1, 32);
          part += __shfl_xor(part, 2, 32);
          part += __shfl_xor(part, 4, 32);
          part += __shfl_xor(part, 8, 32);
          if (ncl == 0) atomicAdd(&trred[mbase + r], part);
        }
      }

      // ---------------- layer 3: dx = H2 @ W3 + b3  (M=16,N=64,K=512) ------
      if (wave < 4) {
        const int ncol = wave * 16 + ncl;
        v8f acc = {};
        for (int kb = 0; kb < H_DIM; kb += 32) {
          v8bf lo = *(const v8bf*)&h2b[arow][kb + koff];
          v8bf hi = *(const v8bf*)&h2b[arow][kb + 16 + koff];
          v16bf a = cat8(lo, hi);
          const v8bf* p = (const v8bf*)&W3T[ncol * H_DIM + kb + kboff];
          acc = wmma_bf16(a, cat8(p[0], p[2]), acc);
        }
#pragma unroll
        for (int r = 0; r < 8; ++r) dxo[mbase + r][ncol] = acc[r] + b3_r;
      }
      __syncthreads();

      // ---------------- RK4 stage combine ----------------------------------
      const float anext = (stage <= 1) ? 0.5f * dt : dt;  // only used stage<3
      for (int e = tid; e < MBLK * D_DIM; e += 256) {
        int m = e >> 6, k = e & 63;
        float dx = dxo[m][k];
        float s  = (stage == 0) ? wgt * dx : xacc[m][k] + wgt * dx;
        if (stage < 3) {
          xacc[m][k] = s;
          xsb[m][k]  = (__bf16)(xcur[m][k] + anext * dx);
        } else {
          float xn = xcur[m][k] + (dt / 6.f) * s;
          xcur[m][k] = xn;
          xsb[m][k]  = (__bf16)xn;
        }
      }
      if (tid < MBLK) {
        float ktr = -trred[tid];
        trred[tid] = 0.f;
        float s = (stage == 0) ? wgt * ktr : ldacc[tid] + wgt * ktr;
        if (stage < 3) ldacc[tid] = s;
        else           ldcur[tid] += (dt / 6.f) * s;
      }
      __syncthreads();
    }
  }

  for (int e = tid; e < MBLK * D_DIM; e += 256) {
    int m = e >> 6, k = e & 63;
    z_out[(rowbase + m) * D_DIM + k] = xcur[m][k];
  }
  if (tid < MBLK) ld_out[rowbase + tid] = ldcur[tid];
}

// ---------------------------------------------------------------------------
extern "C" void kernel_launch(void* const* d_in, const int* in_sizes, int n_in,
                              void* d_out, int out_size, void* d_ws, size_t ws_size,
                              hipStream_t stream) {
  const float* x  = (const float*)d_in[0];
  const float* W1 = (const float*)d_in[1];
  const float* b1 = (const float*)d_in[2];
  const float* W2 = (const float*)d_in[3];
  const float* b2 = (const float*)d_in[4];
  const float* W3 = (const float*)d_in[5];
  const float* b3 = (const float*)d_in[6];

  float* z  = (float*)d_out;
  float* ld = z + 512 * 64;

  __bf16* W1xT = (__bf16*)d_ws;                 // 512*64
  __bf16* W3T  = W1xT + H_DIM * D_DIM;          // 64*512
  __bf16* W2T  = W3T + D_DIM * H_DIM;           // 512*512
  __bf16* ET   = W2T + H_DIM * H_DIM;           // 512*512

  cnf_prep_small<<<(H_DIM * D_DIM + 255) / 256, 256, 0, stream>>>(W1, W3, W1xT, W3T);
  cnf_prep_e<<<H_DIM, H_DIM, 0, stream>>>(W1, W2, W3, W2T, ET);
  cnf_main<<<512 / MBLK, 256, 0, stream>>>(x, W1, b1, b2, b3,
                                           W1xT, W2T, ET, W3T, z, ld);
}
